// S4Naive_59459527246266
// MI455X (gfx1250) — compile-verified, matrix operand order checked by reading, not run on hardware
//
#include <hip/hip_runtime.h>
#include <hip/hip_bf16.h>
#include <hip/hip_fp16.h>

// S4 naive forward for MI455X (gfx1250, wave32, WMMA + TDM).
//
// Pipeline (all on `stream`, sequential):
//   1. s4_prep    : Am = inv(I - A/2), A1 = Am(I + A/2), R = A1^64   (1 block)
//   2. s4_b1      : B1h = f16( Am @ (DELTA*B) )   [64 x 4096]
//   3. s4_gemm_v  : v = x @ B1^T via v_wmma_f32_16x16x32_f16  (dominant, 128 MB stream)
//   4. s4_scan    : per-chunk Horner  s_k = sum_j A1^j v[k*64+j]  (TDM-staged v chunk)
//   5. s4_combine : S = sum_k R^k s_k  via Horner with R = A1^64   (128 steps, 1 block)
//   6. s4_final   : y = C @ S

typedef __attribute__((ext_vector_type(16))) _Float16 v16h;
typedef __attribute__((ext_vector_type(8)))  _Float16 v8h;
typedef __attribute__((ext_vector_type(8)))  float    v8f;
typedef __attribute__((ext_vector_type(4)))  unsigned int u32x4;
typedef __attribute__((ext_vector_type(8)))  int          i32x8;
typedef __attribute__((ext_vector_type(4)))  int          i32x4;

#define NST   64
#define PDIM  4096
#define LSEQ  8192
#define CHUNK 64
#define NCHUNK (LSEQ / CHUNK)   // 128
#define DELTA_C 1.0f

#if __has_builtin(__builtin_amdgcn_tensor_load_to_lds) && \
    __has_builtin(__builtin_amdgcn_s_wait_tensorcnt)
#define S4_USE_TDM 1
#else
#define S4_USE_TDM 0
#endif

// ---------------------------------------------------------------------------
// Kernel 1: discretization setup. Gauss-Jordan on [I - d/2 A | I] in LDS,
// then A1 = Am @ (I + d/2 A), then R = A1^64 by 6 squarings (reusing LDS).
// ---------------------------------------------------------------------------
__global__ __launch_bounds__(256) void s4_prep(const float* __restrict__ A,
                                               float* __restrict__ Am_out,
                                               float* __restrict__ A1_out,
                                               float* __restrict__ R_out) {
    __shared__ float W[NST * 128];    // 32 KB: [I - d/2 A | I]; later 2x 64x64
    __shared__ float A1s[NST * NST];  // 16 KB
    __shared__ float fcol[NST];
    const int tid = threadIdx.x;

    for (int i = tid; i < NST * 128; i += 256) {
        const int r = i >> 7, c = i & 127;
        float val;
        if (c < NST) val = (r == c ? 1.0f : 0.0f) - 0.5f * DELTA_C * A[r * NST + c];
        else         val = (c - NST == r) ? 1.0f : 0.0f;
        W[i] = val;
    }
    __syncthreads();

    // Gauss-Jordan (no pivoting; I - d/2 A_hippo is well conditioned)
    for (int p = 0; p < NST; ++p) {
        const float invp = 1.0f / W[p * 128 + p];
        __syncthreads();
        if (tid < 128) W[p * 128 + tid] *= invp;
        __syncthreads();
        if (tid < NST) fcol[tid] = W[tid * 128 + p];
        __syncthreads();
        for (int i = tid; i < NST * 128; i += 256) {
            const int r = i >> 7;
            if (r != p) W[i] -= fcol[r] * W[p * 128 + (i & 127)];
        }
        __syncthreads();
    }

    // Am = right half; A1 = Am @ (I + d/2 A)
    for (int i = tid; i < NST * NST; i += 256) {
        const int r = i >> 6, m = i & 63;
        Am_out[i] = W[r * 128 + 64 + m];
    }
    for (int i = tid; i < NST * NST; i += 256) {
        const int r = i >> 6, c = i & 63;
        float acc = 0.0f;
        for (int m = 0; m < NST; ++m) {
            const float rhs = (m == c ? 1.0f : 0.0f) + 0.5f * DELTA_C * A[m * NST + c];
            acc += W[r * 128 + 64 + m] * rhs;
        }
        A1s[i] = acc;
        A1_out[i] = acc;
    }
    __syncthreads();

    // R = A1^64 via 6 squarings, reusing W as two 64x64 buffers
    float* bufA = W;
    float* bufB = W + NST * NST;
    for (int i = tid; i < NST * NST; i += 256) bufA[i] = A1s[i];
    __syncthreads();
    for (int it = 0; it < 6; ++it) {
        for (int i = tid; i < NST * NST; i += 256) {
            const int r = i >> 6, c = i & 63;
            float acc = 0.0f;
            for (int m = 0; m < NST; ++m) acc += bufA[r * NST + m] * bufA[m * NST + c];
            bufB[i] = acc;
        }
        __syncthreads();
        float* t2 = bufA; bufA = bufB; bufB = t2;
    }
    for (int i = tid; i < NST * NST; i += 256) R_out[i] = bufA[i];
}

// ---------------------------------------------------------------------------
// Kernel 2: B1h = f16( Am @ (DELTA*B) ), [64 x 4096] row-major in f16.
// ---------------------------------------------------------------------------
__global__ __launch_bounds__(256) void s4_b1(const float* __restrict__ Am,
                                             const float* __restrict__ B,
                                             _Float16* __restrict__ B1h) {
    __shared__ float Ams[NST * NST];
    const int tid = threadIdx.x;
    for (int i = tid; i < NST * NST; i += 256) Ams[i] = Am[i];
    __syncthreads();
    const int p = blockIdx.x * 256 + tid;
    float bc[NST];
#pragma unroll
    for (int m = 0; m < NST; ++m) bc[m] = B[(size_t)m * PDIM + p];
#pragma unroll 1
    for (int n = 0; n < NST; ++n) {
        float acc = 0.0f;
#pragma unroll
        for (int m = 0; m < NST; ++m) acc += Ams[n * NST + m] * bc[m];
        B1h[(size_t)n * PDIM + p] = (_Float16)(acc * DELTA_C);
    }
}

// ---------------------------------------------------------------------------
// Kernel 3: v = x @ B1^T  (M=8192, N=64, K=4096) via WMMA f16 -> f32.
// 512 blocks x 8 waves; each wave: one 16-row M tile, 512-wide K slice,
// 4 N tiles of 16; LDS reduction over the 8 K-split waves.
//
// A frag (16-bit 16x32, ISA 7.12.2): lane m = lane&15, hs = lane>>4,
//   halves h=0..7 -> K = 8*hs + h; h=8..15 -> K = 16 + 8*hs + (h-8).
// B frag mirrored with lane = N column. C/D: lane n = lane&15, VGPR r -> M = r+8*hs.
// ---------------------------------------------------------------------------
__global__ __launch_bounds__(256) void s4_gemm_v(const float* __restrict__ x,
                                                 const _Float16* __restrict__ B1h,
                                                 float* __restrict__ v) {
    __shared__ float red[8 * 16 * 64];   // 32 KB
    const int tid  = threadIdx.x;
    const int lane = tid & 31;
    const int ks   = tid >> 5;           // K-split wave 0..7
    const int m    = lane & 15;
    const int hs   = lane >> 4;          // 0/1
    const int row0 = blockIdx.x * 16;

    const float* xrow = x + (size_t)(row0 + m) * PDIM;
    const int kbeg = ks * (PDIM / 8);    // 512-wide slice

    v8f acc[4] = {};

    for (int kk = 0; kk < PDIM / 8; kk += 32) {
        const int k0 = kbeg + kk;
        // prefetch the x stream ~4 iterations (512B per row) ahead
        __builtin_prefetch(xrow + k0 + 128, 0, 0);
        // 16 f32 of x per lane, b128 loads, convert to f16 in regs
        const float4 a0 = *reinterpret_cast<const float4*>(xrow + k0 + 8 * hs);
        const float4 a1 = *reinterpret_cast<const float4*>(xrow + k0 + 8 * hs + 4);
        const float4 a2 = *reinterpret_cast<const float4*>(xrow + k0 + 8 * hs + 16);
        const float4 a3 = *reinterpret_cast<const float4*>(xrow + k0 + 8 * hs + 20);
        v16h Af;
        Af[0]  = (_Float16)a0.x; Af[1]  = (_Float16)a0.y;
        Af[2]  = (_Float16)a0.z; Af[3]  = (_Float16)a0.w;
        Af[4]  = (_Float16)a1.x; Af[5]  = (_Float16)a1.y;
        Af[6]  = (_Float16)a1.z; Af[7]  = (_Float16)a1.w;
        Af[8]  = (_Float16)a2.x; Af[9]  = (_Float16)a2.y;
        Af[10] = (_Float16)a2.z; Af[11] = (_Float16)a2.w;
        Af[12] = (_Float16)a3.x; Af[13] = (_Float16)a3.y;
        Af[14] = (_Float16)a3.z; Af[15] = (_Float16)a3.w;

#pragma unroll
        for (int nt = 0; nt < 4; ++nt) {
            const _Float16* brow = B1h + (size_t)(nt * 16 + m) * PDIM + k0;
            const v8h b0 = *reinterpret_cast<const v8h*>(brow + 8 * hs);
            const v8h b1 = *reinterpret_cast<const v8h*>(brow + 8 * hs + 16);
            v16h Bf;
#pragma unroll
            for (int i = 0; i < 8; ++i) { Bf[i] = b0[i]; Bf[8 + i] = b1[i]; }
            acc[nt] = __builtin_amdgcn_wmma_f32_16x16x32_f16(
                /*neg_a=*/false, Af, /*neg_b=*/false, Bf,
                /*c_mod=*/(short)0, acc[nt], /*reuse_a=*/false, /*reuse_b=*/false);
        }
    }

    // reduce the 8 K-split partial 16x64 tiles
#pragma unroll
    for (int nt = 0; nt < 4; ++nt)
#pragma unroll
        for (int r = 0; r < 8; ++r)
            red[ks * 1024 + (r + 8 * hs) * 64 + nt * 16 + m] = acc[nt][r];
    __syncthreads();
    for (int e = tid; e < 1024; e += 256) {
        float sum = 0.0f;
#pragma unroll
        for (int w = 0; w < 8; ++w) sum += red[w * 1024 + e];
        v[(size_t)(row0 + (e >> 6)) * NST + (e & 63)] = sum;
    }
}

// ---------------------------------------------------------------------------
// Kernel 4: per-chunk reverse Horner. Block k: c=0; for j=63..0: c = A1@c + v[k*64+j].
// The 64x64 f32 chunk of v (16 KB, contiguous) is staged into LDS with one
// Tensor Data Mover DMA (tensor_load_to_lds + s_wait_tensorcnt); A1 is stored
// transposed in LDS so row reads across threads are bank-conflict free.
// ---------------------------------------------------------------------------
__global__ __launch_bounds__(64) void s4_scan(const float* __restrict__ A1,
                                              const float* __restrict__ v,
                                              float* __restrict__ sChunks) {
    __shared__ float A1t[NST * NST];   // A1t[c][r] = A1[r][c]
    __shared__ float vt[CHUNK * NST];  // staged v chunk (16 KB)
    __shared__ float cb[2][NST];
    const int t = threadIdx.x;

#if S4_USE_TDM
    if (threadIdx.x < 32) {  // wave 0 issues the DMA (uniform branch)
        const unsigned long long gaddr =
            (unsigned long long)(uintptr_t)(v + (size_t)blockIdx.x * CHUNK * NST);
        const unsigned int ldsoff = (unsigned int)(uintptr_t)&vt[0];  // LDS byte offset
        // D# group 0: count=1 | lds_addr | global_addr[56:0] | type=2
        u32x4 g0;
        g0[0] = 1u;
        g0[1] = ldsoff;
        g0[2] = (unsigned int)(gaddr & 0xFFFFFFFFull);
        g0[3] = (unsigned int)((gaddr >> 32) & 0x01FFFFFFull) | (2u << 30);
        // D# group 1: data_size=4B; tensor 64 x 8192 (stride 64); tile 64 x 64
        i32x8 g1;
        g1[0] = (int)(2u << 16);          // workgroup_mask=0, data_size=2 (4 bytes)
        g1[1] = (int)((unsigned)NST << 16);   // tensor_dim0[15:0] in bits[63:48]
        g1[2] = (int)((unsigned)LSEQ << 16);  // tensor_dim0 hi=0 | tensor_dim1 lo
        g1[3] = (int)((unsigned)CHUNK << 16); // tensor_dim1 hi=0 | tile_dim0
        g1[4] = (int)(unsigned)CHUNK;         // tile_dim1 | tile_dim2=0
        g1[5] = (int)(unsigned)NST;           // tensor_dim0_stride[31:0]
        g1[6] = 0;                            // stride hi | tensor_dim1_stride lo
        g1[7] = 0;
        const i32x4 gz4 = {0, 0, 0, 0};             // groups 2/3: dims 2..4 unused
        const i32x8 gz8 = {0, 0, 0, 0, 0, 0, 0, 0}; // extra group (6-arg toolchain)
        __builtin_amdgcn_tensor_load_to_lds(g0, g1, gz4, gz4, gz8, 0);
        __builtin_amdgcn_s_wait_tensorcnt(0);
    }
#else
    for (int i = t; i < CHUNK * NST; i += 64)
        vt[i] = v[(size_t)blockIdx.x * CHUNK * NST + i];
#endif

    for (int i = t; i < NST * NST; i += 64) A1t[(i & 63) * NST + (i >> 6)] = A1[i];
    cb[0][t] = 0.0f; cb[1][t] = 0.0f;
    __syncthreads();

    int cur = 0;
    for (int j = CHUNK - 1; j >= 0; --j) {
        float acc = vt[j * NST + t];
#pragma unroll 8
        for (int m = 0; m < NST; ++m) acc += A1t[m * NST + t] * cb[cur][m];
        cb[cur ^ 1][t] = acc;
        __syncthreads();
        cur ^= 1;
    }
    sChunks[blockIdx.x * NST + t] = cb[cur][t];
}

// ---------------------------------------------------------------------------
// Kernel 5: combine chunks: S = 0; for k=127..0: S = R@S + s_k;  (R = A1^64)
// ---------------------------------------------------------------------------
__global__ __launch_bounds__(64) void s4_combine(const float* __restrict__ R,
                                                 const float* __restrict__ sChunks,
                                                 float* __restrict__ s_out) {
    __shared__ float Rt[NST * NST];
    __shared__ float cb[2][NST];
    const int t = threadIdx.x;
    for (int i = t; i < NST * NST; i += 64) Rt[(i & 63) * NST + (i >> 6)] = R[i];
    cb[0][t] = 0.0f; cb[1][t] = 0.0f;
    __syncthreads();
    int cur = 0;
    for (int k = NCHUNK - 1; k >= 0; --k) {
        float acc = sChunks[k * NST + t];
#pragma unroll 8
        for (int m = 0; m < NST; ++m) acc += Rt[m * NST + t] * cb[cur][m];
        cb[cur ^ 1][t] = acc;
        __syncthreads();
        cur ^= 1;
    }
    s_out[t] = cb[cur][t];
}

// ---------------------------------------------------------------------------
// Kernel 6: y = C @ s   (4096 dots of length 64)
// ---------------------------------------------------------------------------
__global__ __launch_bounds__(256) void s4_final(const float* __restrict__ C,
                                                const float* __restrict__ s,
                                                float* __restrict__ y) {
    __shared__ float ss[NST];
    if (threadIdx.x < NST) ss[threadIdx.x] = s[threadIdx.x];
    __syncthreads();
    const int p = blockIdx.x * 256 + threadIdx.x;
    const float* crow = C + (size_t)p * NST;
    float acc = 0.0f;
#pragma unroll
    for (int n = 0; n < NST; ++n) acc += crow[n] * ss[n];
    y[p] = acc;
}

// ---------------------------------------------------------------------------
extern "C" void kernel_launch(void* const* d_in, const int* in_sizes, int n_in,
                              void* d_out, int out_size, void* d_ws, size_t ws_size,
                              hipStream_t stream) {
    (void)in_sizes; (void)n_in; (void)out_size; (void)ws_size;
    const float* x = (const float*)d_in[0];   // [8192,4096]
    const float* A = (const float*)d_in[1];   // [64,64]
    const float* B = (const float*)d_in[2];   // [64,4096]
    const float* C = (const float*)d_in[3];   // [4096,64]
    float* y = (float*)d_out;                 // [4096]

    char* ws = (char*)d_ws;
    float*    Am  = (float*)(ws);                               // 16 KB
    float*    A1  = (float*)(ws + (16u << 10));                 // 16 KB
    float*    R   = (float*)(ws + (32u << 10));                 // 16 KB
    _Float16* B1h = (_Float16*)(ws + (48u << 10));              // 512 KB
    float*    v   = (float*)(ws + (48u << 10) + (512u << 10));  // 2 MB
    float*    sc  = (float*)(ws + (48u << 10) + (512u << 10) + (2048u << 10)); // 32 KB
    float*    s   = (float*)(ws + (48u << 10) + (512u << 10) + (2048u << 10) + (32u << 10));

    s4_prep   <<<1,            256, 0, stream>>>(A, Am, A1, R);
    s4_b1     <<<PDIM / 256,   256, 0, stream>>>(Am, B, B1h);
    s4_gemm_v <<<LSEQ / 16,    256, 0, stream>>>(x, B1h, v);
    s4_scan   <<<NCHUNK,       64,  0, stream>>>(A1, v, sc);
    s4_combine<<<1,            64,  0, stream>>>(R, sc, s);
    s4_final  <<<PDIM / 256,   256, 0, stream>>>(C, s, y);
}
